// multiheadAttention_31825707663936
// MI455X (gfx1250) — matrix operand order repaired and checked by reading, not compiled, not run on hardware
//
#include <hip/hip_runtime.h>
#include <hip/hip_bf16.h>

typedef __bf16 bf16;
typedef __attribute__((ext_vector_type(16))) __bf16 v16bf;
typedef __attribute__((ext_vector_type(8)))  __bf16 v8bf;
typedef __attribute__((ext_vector_type(8)))  float  v8f;

#define B_    2
#define S_    2048
#define D_    1024
#define H_    16
#define HD_   64
#define INV_SCALE 0.125f
#define NEGINF (-1e30f)

// ---------------------------------------------------------------------------
// CDNA5 async global->LDS copy (ASYNCcnt-tracked), 16B per lane.
// vdst = LDS byte address (low 32 bits of generic shared pointer),
// vaddr = per-lane unsigned byte offset, saddr = uniform 64-bit base.
// ---------------------------------------------------------------------------
__device__ __forceinline__ void async_b128(const void* base, unsigned voff, void* lds) {
  unsigned ldsa = (unsigned)(unsigned long long)lds;
  unsigned long long b = (unsigned long long)base;
  asm volatile("global_load_async_to_lds_b128 %0, %1, %2"
               :: "v"(ldsa), "v"(voff), "s"(b) : "memory");
}
__device__ __forceinline__ void wait_async0() {
  asm volatile("s_wait_asynccnt 0" ::: "memory");
}

// A-fragment (16x32 bf16): lane holds row M=lane%16; K = {klo..klo+7, 16+klo..16+klo+7}
__device__ __forceinline__ v16bf fragA_ld(const bf16* row, int klo) {
  union { v16bf v; v8bf h[2]; } u;
  u.h[0] = *(const v8bf*)(row + klo);
  u.h[1] = *(const v8bf*)(row + klo + 16);
  return u.v;
}
// B-fragment (32x16 bf16): lane holds col N=lane%16; K = {khalf/2..khalf/2+7, 16+khalf/2..16+khalf/2+7}
__device__ __forceinline__ v16bf fragB_ld(const bf16* row, int khalf) {
  union { v16bf v; v8bf h[2]; } u;
  const bf16* p = row + khalf / 2;
  u.h[0] = *(const v8bf*)(p);
  u.h[1] = *(const v8bf*)(p + 16);
  return u.v;
}
__device__ __forceinline__ v8f wmma_bf16(v16bf a, v16bf b, v8f c) {
  return __builtin_amdgcn_wmma_f32_16x16x32_bf16(false, a, false, b, (short)0, c, false, false);
}

__global__ void cvt_f32_bf16(const float* __restrict__ in, bf16* __restrict__ out, int n) {
  int i = blockIdx.x * blockDim.x + threadIdx.x;
  if (i < n) out[i] = (bf16)in[i];
}

// Transposing convert for DxD weights: out[n][k] = (bf16)in[k][n]
__global__ __launch_bounds__(256)
void cvt_transpose_bf16(const float* __restrict__ in, bf16* __restrict__ out) {
  __shared__ float t[32][33];
  const int bx = blockIdx.x * 32;  // n base
  const int by = blockIdx.y * 32;  // k base
  const int tx = threadIdx.x, ty = threadIdx.y;  // (32, 8)
#pragma unroll
  for (int j = 0; j < 32; j += 8)
    t[ty + j][tx] = in[(size_t)(by + ty + j) * D_ + bx + tx];
  __syncthreads();
#pragma unroll
  for (int j = 0; j < 32; j += 8)
    out[(size_t)(bx + ty + j) * D_ + by + tx] = (bf16)t[tx][ty + j];
}

// ---------------------------------------------------------------------------
// GEMM: C[4096x1024] = A[4096x1024] * Wt^T + bias, Wt stored transposed [n][k].
// MODE 0: bf16 scatter -> [b,h,s,hd]   (Q/K projections)
// MODE 2: bf16 scatter -> [b,h,hd,s]   (V projection, pre-transposed for attn)
// MODE 1: f32 row-major                (output projection)
// 256 threads = 8 waves, 128x128 tile, wave tile 32x64 (2x4 WMMA accums),
// async-to-LDS double-buffered k-steps of 32.
// ---------------------------------------------------------------------------
#define TS 40  // padded LDS k-stride (16B-aligned rows, staggers banks)

template<int MODE>
__global__ __launch_bounds__(256)
void gemm4096(const bf16* __restrict__ A, const bf16* __restrict__ Wt,
              const float* __restrict__ bias, void* __restrict__ outp) {
  __shared__ __align__(16) bf16 As[2][128 * TS];
  __shared__ __align__(16) bf16 Bs[2][128 * TS];  // [n][k]

  const int tid  = threadIdx.x;
  const int lane = tid & 31, wid = tid >> 5;
  const int l16  = lane & 15, half = (lane >> 4) & 1;
  const int klo  = half * 8, khalf = half * 16;
  const int wave_m = wid & 3, wave_n = wid >> 2;  // 4 x 2 wave grid
  const int m0 = blockIdx.y * 128;
  const int n0 = blockIdx.x * 128;

  // per-thread staging slot: 16 contiguous bf16 (32B) of one tile row
  const int lrow = tid >> 1;
  const int lk   = (tid & 1) << 4;
  const unsigned aOff0 = ((unsigned)(m0 + lrow) * D_ + lk) * 2u;
  const unsigned bOff0 = ((unsigned)(n0 + lrow) * D_ + lk) * 2u;

  auto issue = [&](int kk, int buf) {
    unsigned ao = aOff0 + (unsigned)kk * 2u;
    unsigned bo = bOff0 + (unsigned)kk * 2u;
    bf16* la = &As[buf][lrow * TS + lk];
    bf16* lb = &Bs[buf][lrow * TS + lk];
    async_b128(A,  ao,      la);
    async_b128(A,  ao + 16, la + 8);
    async_b128(Wt, bo,      lb);
    async_b128(Wt, bo + 16, lb + 8);
  };

  v8f acc[2][4];
#pragma unroll
  for (int mt = 0; mt < 2; ++mt)
#pragma unroll
    for (int nt = 0; nt < 4; ++nt)
#pragma unroll
      for (int r = 0; r < 8; ++r) acc[mt][nt][r] = 0.0f;

  issue(0, 0);
  int buf = 0;
  for (int kk = 0; kk < D_; kk += 32, buf ^= 1) {
    wait_async0();
    __syncthreads();
    if (kk + 32 < D_) issue(kk + 32, buf ^ 1);

    v16bf af[2], bfr[4];
#pragma unroll
    for (int mt = 0; mt < 2; ++mt)
      af[mt] = fragA_ld(&As[buf][(wave_m * 32 + mt * 16 + l16) * TS], klo);
#pragma unroll
    for (int nt = 0; nt < 4; ++nt)
      bfr[nt] = fragB_ld(&Bs[buf][(wave_n * 64 + nt * 16 + l16) * TS], khalf);
#pragma unroll
    for (int mt = 0; mt < 2; ++mt)
#pragma unroll
      for (int nt = 0; nt < 4; ++nt)
        acc[mt][nt] = wmma_bf16(af[mt], bfr[nt], acc[mt][nt]);
  }

  // epilogue: C/D layout row M = r + 8*half, col N = l16
#pragma unroll
  for (int mt = 0; mt < 2; ++mt) {
#pragma unroll
    for (int nt = 0; nt < 4; ++nt) {
#pragma unroll
      for (int r = 0; r < 8; ++r) {
        int gm = m0 + wave_m * 32 + mt * 16 + r + half * 8;
        int gn = n0 + wave_n * 64 + nt * 16 + l16;
        float v = acc[mt][nt][r] + bias[gn];
        if (MODE == 1) {
          ((float*)outp)[(size_t)gm * D_ + gn] = v;
        } else {
          bf16* ob = (bf16*)outp;
          int b = gm >> 11, s = gm & (S_ - 1);
          int h = gn >> 6,  hd = gn & (HD_ - 1);
          if (MODE == 0)
            ob[(((size_t)b * H_ + h) * S_ + s) * HD_ + hd] = (bf16)v;
          else  // MODE 2: V transposed per head -> [b,h,hd,s]
            ob[(((size_t)b * H_ + h) * HD_ + hd) * S_ + s] = (bf16)v;
        }
      }
    }
  }
}

// ---------------------------------------------------------------------------
// Flash attention: block per (b*h, 64-row query block), 128 threads = 4 waves,
// each wave owns 16 query rows. Async double-buffered K/V tiles; V arrives
// pre-transposed [hd][s] so every stage is a contiguous b128 copy.
// ---------------------------------------------------------------------------
#define AST 72

__global__ __launch_bounds__(128)
void attn_kernel(const bf16* __restrict__ Q, const bf16* __restrict__ Kmat,
                 const bf16* __restrict__ Vt, bf16* __restrict__ ctx) {
  __shared__ __align__(16) bf16 Qs[64 * AST];
  __shared__ __align__(16) bf16 Ks[2][64 * AST];
  __shared__ __align__(16) bf16 Vs[2][64 * AST];   // [hd][key]
  __shared__ __align__(16) bf16 Ps[4 * 16 * AST];  // per-wave P staging

  const int tid  = threadIdx.x;
  const int lane = tid & 31, wid = tid >> 5;
  const int l16  = lane & 15, half = (lane >> 4) & 1;
  const int klo  = half * 8, khalf = half * 16;
  const int qb = blockIdx.x;
  const int bh = blockIdx.y;
  const int b  = bh >> 4, h = bh & 15;

  const bf16* gQ = Q    + (size_t)bh * S_ * HD_ + (size_t)qb * 64 * HD_;
  const bf16* gK = Kmat + (size_t)bh * S_ * HD_;
  const bf16* gV = Vt   + (size_t)bh * HD_ * S_;

  const int lrow = tid >> 1;        // 0..63
  const int c0   = (tid & 1) << 5;  // 0 or 32

  // Q block (64x64): 4 async b128 per thread
  {
    unsigned off = ((unsigned)lrow * HD_ + c0) * 2u;
    bf16* d = &Qs[lrow * AST + c0];
    async_b128(gQ, off,      d);
    async_b128(gQ, off + 16, d + 8);
    async_b128(gQ, off + 32, d + 16);
    async_b128(gQ, off + 48, d + 24);
  }
  auto issueKV = [&](int kb, int bufi) {
    unsigned offK = ((unsigned)(kb * 64 + lrow) * HD_ + c0) * 2u;
    bf16* dK = &Ks[bufi][lrow * AST + c0];
    async_b128(gK, offK,      dK);
    async_b128(gK, offK + 16, dK + 8);
    async_b128(gK, offK + 32, dK + 16);
    async_b128(gK, offK + 48, dK + 24);
    unsigned offV = ((unsigned)lrow * S_ + (unsigned)(kb * 64 + c0)) * 2u;
    bf16* dV = &Vs[bufi][lrow * AST + c0];
    async_b128(gV, offV,      dV);
    async_b128(gV, offV + 16, dV + 8);
    async_b128(gV, offV + 32, dV + 16);
    async_b128(gV, offV + 48, dV + 24);
  };

  issueKV(0, 0);
  wait_async0();
  __syncthreads();

  v16bf qf[2];
#pragma unroll
  for (int kk = 0; kk < 2; ++kk)
    qf[kk] = fragA_ld(&Qs[(wid * 16 + l16) * AST + kk * 32], klo);

  float m[8], l[8];
  v8f O[4];
#pragma unroll
  for (int r = 0; r < 8; ++r) { m[r] = NEGINF; l[r] = 0.0f; }
#pragma unroll
  for (int nt = 0; nt < 4; ++nt)
#pragma unroll
    for (int r = 0; r < 8; ++r) O[nt][r] = 0.0f;

  bf16* Pw = &Ps[wid * 16 * AST];

  int buf = 0;
  for (int kb = 0; kb <= qb; ++kb, buf ^= 1) {
    if (kb > 0) { wait_async0(); __syncthreads(); }
    if (kb < qb) issueKV(kb + 1, buf ^ 1);

    // scores: S(16x64) = Q(16x64) x K^T(64x64)
    v8f sc[4];
#pragma unroll
    for (int nt = 0; nt < 4; ++nt) {
#pragma unroll
      for (int r = 0; r < 8; ++r) sc[nt][r] = 0.0f;
      v16bf bk0 = fragB_ld(&Ks[buf][(nt * 16 + l16) * AST + 0],  khalf);
      v16bf bk1 = fragB_ld(&Ks[buf][(nt * 16 + l16) * AST + 32], khalf);
      sc[nt] = wmma_bf16(qf[0], bk0, sc[nt]);
      sc[nt] = wmma_bf16(qf[1], bk1, sc[nt]);
    }

    // online softmax per query row (row lives across a 16-lane group)
#pragma unroll
    for (int r = 0; r < 8; ++r) {
      int qrow = qb * 64 + wid * 16 + r + half * 8;
#pragma unroll
      for (int nt = 0; nt < 4; ++nt) {
        int kcol = kb * 64 + nt * 16 + l16;
        float v = sc[nt][r] * INV_SCALE;
        sc[nt][r] = (kcol <= qrow) ? v : NEGINF;
      }
      float rm = fmaxf(fmaxf(sc[0][r], sc[1][r]), fmaxf(sc[2][r], sc[3][r]));
      rm = fmaxf(rm, __shfl_xor(rm, 1, 32));
      rm = fmaxf(rm, __shfl_xor(rm, 2, 32));
      rm = fmaxf(rm, __shfl_xor(rm, 4, 32));
      rm = fmaxf(rm, __shfl_xor(rm, 8, 32));
      float mn = fmaxf(m[r], rm);
      float al = __expf(m[r] - mn);
      float ls = 0.0f;
#pragma unroll
      for (int nt = 0; nt < 4; ++nt) {
        float p = __expf(sc[nt][r] - mn);
        sc[nt][r] = p;
        ls += p;
      }
      ls += __shfl_xor(ls, 1, 32);
      ls += __shfl_xor(ls, 2, 32);
      ls += __shfl_xor(ls, 4, 32);
      ls += __shfl_xor(ls, 8, 32);
      l[r] = l[r] * al + ls;
      m[r] = mn;
#pragma unroll
      for (int nt = 0; nt < 4; ++nt) O[nt][r] *= al;
    }

    // stage P (bf16) to LDS row-major [row16][key64] for A-fragment reload
#pragma unroll
    for (int r = 0; r < 8; ++r)
#pragma unroll
      for (int nt = 0; nt < 4; ++nt)
        Pw[(r + half * 8) * AST + nt * 16 + l16] = (bf16)sc[nt][r];
    __syncthreads();

    // O(16x64) += P(16x64) x V(64x64)
    v16bf pf0 = fragA_ld(&Pw[l16 * AST + 0],  klo);
    v16bf pf1 = fragA_ld(&Pw[l16 * AST + 32], klo);
#pragma unroll
    for (int nt = 0; nt < 4; ++nt) {
      v16bf bv0 = fragB_ld(&Vs[buf][(nt * 16 + l16) * AST + 0],  khalf);
      v16bf bv1 = fragB_ld(&Vs[buf][(nt * 16 + l16) * AST + 32], khalf);
      O[nt] = wmma_bf16(pf0, bv0, O[nt]);
      O[nt] = wmma_bf16(pf1, bv1, O[nt]);
    }
  }

  // normalize and write ctx as bf16 in [B*S, D] layout (merging heads)
#pragma unroll
  for (int nt = 0; nt < 4; ++nt)
#pragma unroll
    for (int r = 0; r < 8; ++r) {
      int qrow = qb * 64 + wid * 16 + r + half * 8;
      int hd = nt * 16 + l16;
      float v = O[nt][r] / l[r];
      ctx[((size_t)b * S_ + qrow) * D_ + h * HD_ + hd] = (bf16)v;
    }
}

// ---------------------------------------------------------------------------
extern "C" void kernel_launch(void* const* d_in, const int* in_sizes, int n_in,
                              void* d_out, int out_size, void* d_ws, size_t ws_size,
                              hipStream_t stream) {
  (void)in_sizes; (void)n_in; (void)out_size; (void)ws_size;
  const float* x  = (const float*)d_in[0];
  const float* Wq = (const float*)d_in[1];
  const float* bq = (const float*)d_in[2];
  const float* Wk = (const float*)d_in[3];
  const float* bk = (const float*)d_in[4];
  const float* Wv = (const float*)d_in[5];
  const float* bv = (const float*)d_in[6];
  const float* Wo = (const float*)d_in[7];
  const float* bo = (const float*)d_in[8];
  float* out = (float*)d_out;

  char* ws = (char*)d_ws;
  size_t off = 0;
  auto alloc = [&](size_t bytes) {
    char* p = ws + off;
    off += (bytes + 255) & ~(size_t)255;
    return p;
  };
  const size_t nx = (size_t)B_ * S_ * D_;  // 4194304
  const size_t nw = (size_t)D_ * D_;       // 1048576
  bf16* xb   = (bf16*)alloc(nx * 2);
  bf16* Wqt  = (bf16*)alloc(nw * 2);
  bf16* Wkt  = (bf16*)alloc(nw * 2);
  bf16* Wvt  = (bf16*)alloc(nw * 2);
  bf16* Wot  = (bf16*)alloc(nw * 2);
  bf16* Qb   = (bf16*)alloc(nx * 2);
  bf16* Kb   = (bf16*)alloc(nx * 2);
  bf16* Vtb  = (bf16*)alloc(nx * 2);
  bf16* ctxb = (bf16*)alloc(nx * 2);

  cvt_f32_bf16<<<(unsigned)((nx + 255) / 256), 256, 0, stream>>>(x, xb, (int)nx);
  dim3 tb(32, 8), tg(D_ / 32, D_ / 32);
  cvt_transpose_bf16<<<tg, tb, 0, stream>>>(Wq, Wqt);
  cvt_transpose_bf16<<<tg, tb, 0, stream>>>(Wk, Wkt);
  cvt_transpose_bf16<<<tg, tb, 0, stream>>>(Wv, Wvt);
  cvt_transpose_bf16<<<tg, tb, 0, stream>>>(Wo, Wot);

  dim3 gg(D_ / 128, (B_ * S_) / 128);  // (8, 32)
  gemm4096<0><<<gg, 256, 0, stream>>>(xb, Wqt, bq, (void*)Qb);
  gemm4096<0><<<gg, 256, 0, stream>>>(xb, Wkt, bk, (void*)Kb);
  gemm4096<2><<<gg, 256, 0, stream>>>(xb, Wvt, bv, (void*)Vtb);

  dim3 ga(S_ / 64, B_ * H_);  // (32, 32)
  attn_kernel<<<ga, 128, 0, stream>>>(Qb, Kb, Vtb, ctxb);

  gemm4096<1><<<gg, 256, 0, stream>>>(ctxb, Wot, bo, (void*)out);
}
